// ExpDock_13271448944739
// MI455X (gfx1250) — matrix operand order repaired and checked by reading, not compiled, not run on hardware
//
#include <hip/hip_runtime.h>
#include <hip/hip_bf16.h>

#define NN    16384
#define EE    147456
#define KD    9
#define HH    128
#define LL    4
#define RBF_D 20
#define ED_D  16
#define NSEG  32
#define SEGSZ 512
#define PI_F  3.14159265358979f

typedef __attribute__((ext_vector_type(16))) _Float16 v16h;
typedef __attribute__((ext_vector_type(8)))  float    v8f;

// ---------------- weight packing: f32 row-major [K][Nw] -> f16 WMMA B tiles ----------------
// Packed element index: (((nt*Kt)+kt)*32 + lane)*16 + j
// B layout (16-bit, 32x16 per tile): lane holds column n = nt*16 + (lane&15),
// halves j=0..15 hold k = kt*32 + (lane>>4)*8 + (j&7) + (j>>3)*16.
__global__ void pack_w_kernel(const float* __restrict__ W, _Float16* __restrict__ out,
                              int K, int Nw, int Kt) {
  int t = blockIdx.x * blockDim.x + threadIdx.x;
  int total = (Nw / 16) * Kt * 512;
  if (t >= total) return;
  int j    = t & 15;
  int lane = (t >> 4) & 31;
  int kt   = (t >> 9) % Kt;
  int nt   = t / (512 * Kt);
  int k = kt * 32 + (lane >> 4) * 8 + (j & 7) + (j >> 3) * 16;
  int n = nt * 16 + (lane & 15);
  float v = (k < K && n < Nw) ? W[(size_t)k * Nw + n] : 0.f;
  out[t] = (_Float16)v;
}

// 8 contiguous f32 -> 8 f16 halves of the A fragment (16B-aligned source)
__device__ __forceinline__ void load8(const float* __restrict__ p, v16h& a, int base) {
  float4 x0 = *(const float4*)(p);
  float4 x1 = *(const float4*)(p + 4);
  a[base + 0] = (_Float16)x0.x; a[base + 1] = (_Float16)x0.y;
  a[base + 2] = (_Float16)x0.z; a[base + 3] = (_Float16)x0.w;
  a[base + 4] = (_Float16)x1.x; a[base + 5] = (_Float16)x1.y;
  a[base + 6] = (_Float16)x1.z; a[base + 7] = (_Float16)x1.w;
}

// tail of the edge-gather row: k in [256, 320) -> rbf[20] | edge_attr[16] | pad
__device__ __forceinline__ float fetch_tail(const float* __restrict__ rbf,
                                            const float* __restrict__ ea, int m, int k) {
  k -= 2 * HH;
  if (k < RBF_D) return rbf[(size_t)m * RBF_D + k];
  k -= RBF_D;
  if (k < ED_D) return ea[(size_t)m * ED_D + k];
  return 0.f;
}

// A fragment (16x32 f16): lane l -> row m = mBase + (l&15);
// halves 0..7 -> k = kt*32 + (l>>4)*8 + j ; halves 8..15 -> k = kt*32 + 16 + (l>>4)*8 + j
// MODE 0: row-major A0[M][K0] (K0 multiple of 32)
// MODE 1: edge gather [ h[e/9] | h[col[e]] | rbf | ea | pad ]  (Kt = 10)
// MODE 2: concat A0[M][K0] ++ A1[M][K1] (both multiples of 32)
template <int MODE>
__device__ __forceinline__ v16h build_afrag(const float* __restrict__ A0,
                                            const float* __restrict__ A1,
                                            const float* __restrict__ rbf,
                                            const float* __restrict__ ea,
                                            const int* __restrict__ colIdx,
                                            int mBase, int lane, int ktBase, int M,
                                            int K0, int K1) {
  v16h a = {};
  int m = mBase + (lane & 15);
  if (m >= M) return a;
  int kh = (lane >> 4) * 8;
  int kA = ktBase + kh;
  int kB = ktBase + 16 + kh;
  if (MODE == 0) {
    const float* row = A0 + (size_t)m * K0;
    load8(row + kA, a, 0);
    load8(row + kB, a, 8);
  } else if (MODE == 2) {
    const float* rowlo = A0 + (size_t)m * K0;
    const float* rowhi = A1 + (size_t)m * K1;
    const float* p1 = (kA < K0) ? (rowlo + kA) : (rowhi + (kA - K0));
    const float* p2 = (kB < K0) ? (rowlo + kB) : (rowhi + (kB - K0));
    load8(p1, a, 0);
    load8(p2, a, 8);
  } else {
    if (ktBase < 2 * HH) {  // uniform scalar branch: whole tile inside h[row]|h[col]
      const float* rowr = A0 + (size_t)(m / KD) * HH;
      const float* rowc = A0 + (size_t)colIdx[m] * HH;
      const float* p1 = (kA < HH) ? (rowr + kA) : (rowc + (kA - HH));
      const float* p2 = (kB < HH) ? (rowr + kB) : (rowc + (kB - HH));
      load8(p1, a, 0);
      load8(p2, a, 8);
    } else {  // rbf/edge_attr tail (tiles 8..9)
#pragma unroll
      for (int j = 0; j < 8; ++j) a[j] = (_Float16)fetch_tail(rbf, ea, m, kA + j);
#pragma unroll
      for (int j = 0; j < 8; ++j) a[8 + j] = (_Float16)fetch_tail(rbf, ea, m, kB + j);
    }
  }
  return a;
}

// ---------------- generic WMMA GEMM: C[M][Nw] = act(A@B + bias) (+resid) ----------------
// One wave per block computes a 16x64 output strip (4 column tiles), K looped in steps of 32.
template <int MODE>
__global__ __launch_bounds__(32) void gemm_wmma_kernel(
    const float* __restrict__ A0, const float* __restrict__ A1,
    const float* __restrict__ rbf, const float* __restrict__ ea,
    const int* __restrict__ colIdx, const _Float16* __restrict__ Bp,
    const float* __restrict__ bias, const float* __restrict__ resid,
    float* __restrict__ C, int M, int K0, int K1, int Kt, int Nw, int act) {
  int lane  = threadIdx.x;
  int mBase = blockIdx.x * 16;
  int nt0   = blockIdx.y * 4;

  v8f acc[4] = {};
  for (int kt = 0; kt < Kt; ++kt) {
    v16h a = build_afrag<MODE>(A0, A1, rbf, ea, colIdx, mBase, lane, kt * 32, M, K0, K1);
#pragma unroll
    for (int t = 0; t < 4; ++t) {
      const _Float16* bp = Bp + (((size_t)(nt0 + t) * Kt + kt) * 512) + (size_t)lane * 16;
      v16h b = *(const v16h*)bp;
      acc[t] = __builtin_amdgcn_wmma_f32_16x16x32_f16(false, a, false, b, (short)0, acc[t],
                                                      false, false);
    }
  }

  int half  = lane >> 4;
  int nlane = lane & 15;
#pragma unroll
  for (int t = 0; t < 4; ++t) {
    int n = (nt0 + t) * 16 + nlane;
#pragma unroll
    for (int r = 0; r < 8; ++r) {
      int m = mBase + r + 8 * half;
      if (m < M) {
        float x = acc[t][r];
        if (bias)  x += bias[n];
        if (act)   x = x / (1.f + __expf(-x));   // silu
        if (resid) x += resid[(size_t)m * Nw + n];
        C[(size_t)m * Nw + n] = x;
      }
    }
  }
}

// ---------------- elementwise / reduction kernels ----------------
__global__ void zero_kernel(float* p, int n) {
  int t = blockIdx.x * blockDim.x + threadIdx.x;
  if (t < n) p[t] = 0.f;
}

__global__ void geom_kernel(const float* __restrict__ X, const int* __restrict__ colIdx,
                            float* __restrict__ unit, float* __restrict__ rbf) {
  int e = blockIdx.x * blockDim.x + threadIdx.x;
  if (e >= EE) return;
  int r = e / KD, c = colIdx[e];
  float dx = X[r * 3 + 0] - X[c * 3 + 0];
  float dy = X[r * 3 + 1] - X[c * 3 + 1];
  float dz = X[r * 3 + 2] - X[c * 3 + 2];
  float norm = sqrtf(dx * dx + dy * dy + dz * dz) + 1e-8f;
  float inv = 1.f / norm;
  unit[e * 3 + 0] = dx * inv;
  unit[e * 3 + 1] = dy * inv;
  unit[e * 3 + 2] = dz * inv;
  float cl  = fminf(fmaxf(norm, 0.f), 1.0f);
  float cut = 0.5f * (__cosf(PI_F * cl) + 1.f);
#pragma unroll
  for (int i = 0; i < RBF_D; ++i) {
    float fr = (float)(i + 1) * PI_F;
    rbf[(size_t)e * RBF_D + i] = __sinf(norm * fr) * inv * cut;
  }
}

// agg[n][f] = sum_{j<9} m[n*9+j][f]
__global__ void agg_kernel(const float* __restrict__ m, float* __restrict__ agg) {
  int t = blockIdx.x * blockDim.x + threadIdx.x;
  if (t >= NN * HH) return;
  int n = t / HH, f = t % HH;
  float acc = 0.f;
#pragma unroll
  for (int j = 0; j < KD; ++j) acc += m[(size_t)(n * KD + j) * HH + f];
  agg[t] = acc;
}

// v stored [n][k][c] (c fastest). vnew = vold + sum_e mv[e][k]*unit[e][c] + mvv[e][k]*vold[col[e]][k][c]
__global__ void v_update_kernel(const float* __restrict__ vold, const float* __restrict__ mv,
                                const float* __restrict__ mvv, const float* __restrict__ unit,
                                const int* __restrict__ colIdx, float* __restrict__ vnew) {
  int t = blockIdx.x * blockDim.x + threadIdx.x;
  if (t >= NN * HH) return;
  int n = t / HH, k = t % HH;
  size_t base = (size_t)n * (HH * 3) + (size_t)k * 3;
  float a0 = vold[base + 0], a1 = vold[base + 1], a2 = vold[base + 2];
#pragma unroll
  for (int j = 0; j < KD; ++j) {
    int e = n * KD + j;
    float s1 = mv[(size_t)e * HH + k];
    float s2 = mvv[(size_t)e * HH + k];
    int c = colIdx[e];
    size_t cb = (size_t)c * (HH * 3) + (size_t)k * 3;
    a0 += s1 * unit[e * 3 + 0] + s2 * vold[cb + 0];
    a1 += s1 * unit[e * 3 + 1] + s2 * vold[cb + 1];
    a2 += s1 * unit[e * 3 + 2] + s2 * vold[cb + 2];
  }
  vnew[base + 0] = a0; vnew[base + 1] = a1; vnew[base + 2] = a2;
}

// per-segment feature sums over contiguous 512-node blocks
__global__ void seg_sum_kernel(const float* __restrict__ h, float* __restrict__ out) {
  int s = blockIdx.x, f = threadIdx.x;
  float acc = 0.f;
  int base = s * SEGSZ;
  for (int i = 0; i < SEGSZ; ++i) acc += h[(size_t)(base + i) * HH + f];
  out[s * HH + f] = acc;
}

__global__ void score_kernel(const float* __restrict__ hi, const float* __restrict__ segsum,
                             float* __restrict__ score) {
  int n = blockIdx.x * blockDim.x + threadIdx.x;
  if (n >= NN) return;
  int opp = (n / SEGSZ) ^ 1;
  float acc = 0.f;
  for (int f = 0; f < HH; ++f) acc += hi[(size_t)n * HH + f] * segsum[opp * HH + f];
  score[n] = acc * (1.f / (float)SEGSZ);
}

__global__ void softstat_kernel(const float* __restrict__ score, float* __restrict__ smax,
                                float* __restrict__ sumex) {
  __shared__ float sh[256];
  int s = blockIdx.x, t = threadIdx.x;
  float m = -1e30f;
  for (int i = t; i < SEGSZ; i += 256) m = fmaxf(m, score[s * SEGSZ + i]);
  sh[t] = m; __syncthreads();
  for (int o = 128; o > 0; o >>= 1) { if (t < o) sh[t] = fmaxf(sh[t], sh[t + o]); __syncthreads(); }
  float mx = sh[0]; __syncthreads();
  float su = 0.f;
  for (int i = t; i < SEGSZ; i += 256) su += __expf(score[s * SEGSZ + i] - mx);
  sh[t] = su; __syncthreads();
  for (int o = 128; o > 0; o >>= 1) { if (t < o) sh[t] += sh[t + o]; __syncthreads(); }
  if (t == 0) { smax[s] = mx; sumex[s] = sh[0]; }
}

__global__ void gated_kernel(const float* __restrict__ hi, const float* __restrict__ score,
                             const float* __restrict__ smax, const float* __restrict__ sumex,
                             float* __restrict__ gated) {
  int t = blockIdx.x * blockDim.x + threadIdx.x;
  if (t >= NN * HH) return;
  int n = t / HH;
  int s = n / SEGSZ;
  float w = __expf(score[n] - smax[s]) / sumex[s];
  gated[t] = hi[t] * w;
}

// vt[(n*3+c)*128 + k] = v[n][k][c]
__global__ void vt_kernel(const float* __restrict__ v, float* __restrict__ vt) {
  int t = blockIdx.x * blockDim.x + threadIdx.x;
  if (t >= NN * HH * 3) return;
  int n = t / (HH * 3);
  int rem = t % (HH * 3);
  int k = rem / 3, c = rem % 3;
  vt[((size_t)(n * 3 + c)) * HH + k] = v[t];
}

__global__ void v2n_kernel(const float* __restrict__ v2t, float* __restrict__ v2n) {
  int t = blockIdx.x * blockDim.x + threadIdx.x;
  if (t >= NN * HH) return;
  int n = t / HH, k = t % HH;
  float s = 0.f;
#pragma unroll
  for (int c = 0; c < 3; ++c) {
    float x = v2t[((size_t)(n * 3 + c)) * HH + k];
    s += x * x;
  }
  v2n[t] = sqrtf(s + 1e-8f);
}

// pv[n][o*3+c] = sum_k v1t[(n*3+c)*128+k] * bgate[n][k] * hnew[n][k] * Wf[k][o]
__global__ void pv_kernel(const float* __restrict__ v1t, const float* __restrict__ mix,
                          const float* __restrict__ Wf, float* __restrict__ pv) {
  int t = blockIdx.x * blockDim.x + threadIdx.x;
  if (t >= NN * 27) return;
  int n = t / 27, rem = t % 27;
  int o = rem / 3, c = rem % 3;
  const float* vr = v1t + ((size_t)(n * 3 + c)) * HH;
  const float* mr = mix + (size_t)n * 256;
  float acc = 0.f;
  for (int k = 0; k < HH; ++k)
    acc += vr[k] * mr[128 + k] * mr[k] * Wf[k * 9 + o];
  pv[t] = acc;
}

__global__ void pred_kernel(const float* __restrict__ pv, float* __restrict__ pred) {
  int s = blockIdx.x, r = threadIdx.x;  // 32 blocks x 27 threads
  float acc = 0.f;
  int base = s * SEGSZ;
  for (int i = 0; i < SEGSZ; ++i) acc += pv[(size_t)(base + i) * 27 + r];
  pred[s * 27 + r] = acc;
}

__global__ void out_kernel(const float* __restrict__ pred, float* __restrict__ out) {
  int s = blockIdx.x, t = threadIdx.x;  // 32 blocks x 12 threads
  int i = t / 4, j = t % 4;
  const float* p = pred + s * 27;       // p[o*3 + x]
  float val;
  if (j < 3)
    val = (p[0 * 3 + i] * p[1 * 3 + j] + p[3 * 3 + i] * p[2 * 3 + j] +
           p[4 * 3 + i] * p[5 * 3 + j] + p[7 * 3 + i] * p[6 * 3 + j]) * 100.f;  // 1/SF^2
  else
    val = p[8 * 3 + i] * 10.f;  // 1/SF
  out[s * 12 + i * 4 + j] = val;
}

// ---------------- host launcher ----------------
extern "C" void kernel_launch(void* const* d_in, const int* in_sizes, int n_in,
                              void* d_out, int out_size, void* d_ws, size_t ws_size,
                              hipStream_t stream) {
  (void)in_sizes; (void)n_in; (void)out_size; (void)ws_size;
  const float* X         = (const float*)d_in[0];
  const float* node_attr = (const float*)d_in[1];
  const float* edge_attr = (const float*)d_in[2];
  const float* W_in1 = (const float*)d_in[3];
  const float* b_in1 = (const float*)d_in[4];
  const float* W_in2 = (const float*)d_in[5];
  const float* b_in2 = (const float*)d_in[6];
  const float* We1 = (const float*)d_in[7];
  const float* be1 = (const float*)d_in[8];
  const float* We2 = (const float*)d_in[9];
  const float* be2 = (const float*)d_in[10];
  const float* Wv  = (const float*)d_in[11];
  const float* Wvv = (const float*)d_in[12];
  const float* Wh1 = (const float*)d_in[13];
  const float* bh1 = (const float*)d_in[14];
  const float* Wh2 = (const float*)d_in[15];
  const float* bh2 = (const float*)d_in[16];
  const float* Wi1 = (const float*)d_in[17];
  const float* bi1 = (const float*)d_in[18];
  const float* Wi2 = (const float*)d_in[19];
  const float* bi2 = (const float*)d_in[20];
  const float* Wgv1 = (const float*)d_in[21];
  const float* Wgv2 = (const float*)d_in[22];
  const float* Wg1  = (const float*)d_in[23];
  const float* bg1  = (const float*)d_in[24];
  const float* Wg2  = (const float*)d_in[25];
  const float* bg2  = (const float*)d_in[26];
  const float* Wf   = (const float*)d_in[27];
  const int*   edges = (const int*)d_in[28];
  const int*   col   = edges + EE;   // edges[0]=row (== e/9), edges[1]=col

  // workspace carve-out
  char* wsp = (char*)d_ws;
  size_t off = 0;
  auto take = [&](size_t bytes) -> void* {
    off = (off + 255) & ~(size_t)255;
    void* p = wsp + off;
    off += bytes;
    return p;
  };
  float* unitB  = (float*)take((size_t)EE * 3 * 4);
  float* rbfB   = (float*)take((size_t)EE * RBF_D * 4);
  float* hB     = (float*)take((size_t)NN * HH * 4);
  float* hxB    = (float*)take((size_t)NN * HH * 4);
  float* aggB   = (float*)take((size_t)NN * HH * 4);
  float* hiB    = (float*)take((size_t)NN * HH * 4);
  float* gatedB = (float*)take((size_t)NN * HH * 4);
  float* segsumB = (float*)take((size_t)NSEG * HH * 4);
  float* scoreB  = (float*)take((size_t)NN * 4);
  float* smaxB   = (float*)take((size_t)NSEG * 4);
  float* sumexB  = (float*)take((size_t)NSEG * 4);
  float* ebuf0 = (float*)take((size_t)EE * HH * 4);
  float* ebuf1 = (float*)take((size_t)EE * HH * 4);
  float* ebuf2 = (float*)take((size_t)EE * HH * 4);
  float* va    = (float*)take((size_t)NN * HH * 3 * 4);
  float* vb    = (float*)take((size_t)NN * HH * 3 * 4);
  float* vtB   = (float*)take((size_t)NN * HH * 3 * 4);
  float* v1tB  = (float*)take((size_t)NN * HH * 3 * 4);
  float* v2tB  = (float*)take((size_t)NN * HH * 3 * 4);
  float* v2nB  = (float*)take((size_t)NN * HH * 4);
  float* mixB  = (float*)take((size_t)NN * 256 * 4);
  float* pvB   = (float*)take((size_t)NN * 27 * 4);
  float* predB = (float*)take((size_t)NSEG * 27 * 4);

  auto psize = [](int Nw, int Kt) -> size_t { return (size_t)(Nw / 16) * Kt * 512; };
  _Float16* pWin1 = (_Float16*)take(psize(128, 2) * 2);
  _Float16* pWin2 = (_Float16*)take(psize(128, 4) * 2);
  _Float16* pWe1  = (_Float16*)take(psize(128, 10) * LL * 2);
  _Float16* pWe2  = (_Float16*)take(psize(128, 4) * LL * 2);
  _Float16* pWv   = (_Float16*)take(psize(128, 4) * LL * 2);
  _Float16* pWvv  = (_Float16*)take(psize(128, 4) * LL * 2);
  _Float16* pWh1  = (_Float16*)take(psize(128, 8) * LL * 2);
  _Float16* pWh2  = (_Float16*)take(psize(128, 4) * LL * 2);
  _Float16* pWi1  = (_Float16*)take(psize(128, 4) * LL * 2);
  _Float16* pWi2  = (_Float16*)take(psize(128, 4) * LL * 2);
  _Float16* pWgv1 = (_Float16*)take(psize(128, 4) * 2);
  _Float16* pWgv2 = (_Float16*)take(psize(128, 4) * 2);
  _Float16* pWg1  = (_Float16*)take(psize(128, 8) * 2);
  _Float16* pWg2  = (_Float16*)take(psize(256, 4) * 2);

  auto pack = [&](const float* W, _Float16* dst, int K, int Nw, int Kt) {
    int total = (Nw / 16) * Kt * 512;
    pack_w_kernel<<<(total + 255) / 256, 256, 0, stream>>>(W, dst, K, Nw, Kt);
  };
  const int MI = 2 * HH + RBF_D + ED_D;  // 292
  pack(W_in1, pWin1, 64, 128, 2);
  pack(W_in2, pWin2, 128, 128, 4);
  for (int i = 0; i < LL; ++i) {
    pack(We1 + (size_t)i * MI * HH,     pWe1 + i * psize(128, 10), MI, 128, 10);
    pack(We2 + (size_t)i * HH * HH,     pWe2 + i * psize(128, 4), 128, 128, 4);
    pack(Wv  + (size_t)i * HH * HH,     pWv  + i * psize(128, 4), 128, 128, 4);
    pack(Wvv + (size_t)i * HH * HH,     pWvv + i * psize(128, 4), 128, 128, 4);
    pack(Wh1 + (size_t)i * 2 * HH * HH, pWh1 + i * psize(128, 8), 256, 128, 8);
    pack(Wh2 + (size_t)i * HH * HH,     pWh2 + i * psize(128, 4), 128, 128, 4);
    pack(Wi1 + (size_t)i * HH * HH,     pWi1 + i * psize(128, 4), 128, 128, 4);
    pack(Wi2 + (size_t)i * HH * HH,     pWi2 + i * psize(128, 4), 128, 128, 4);
  }
  pack(Wgv1, pWgv1, 128, 128, 4);
  pack(Wgv2, pWgv2, 128, 128, 4);
  pack(Wg1,  pWg1,  256, 128, 8);
  pack(Wg2,  pWg2,  128, 256, 4);

  auto gemm = [&](int mode, const float* A0, const float* A1, const float* rbfp,
                  const float* eap, const int* colp, const _Float16* Bp, const float* bias,
                  const float* resid, float* C, int M, int K0, int K1, int Kt, int Nw,
                  int act) {
    dim3 g((M + 15) / 16, Nw / 64);
    if (mode == 0)
      gemm_wmma_kernel<0><<<g, 32, 0, stream>>>(A0, A1, rbfp, eap, colp, Bp, bias, resid, C,
                                                M, K0, K1, Kt, Nw, act);
    else if (mode == 1)
      gemm_wmma_kernel<1><<<g, 32, 0, stream>>>(A0, A1, rbfp, eap, colp, Bp, bias, resid, C,
                                                M, K0, K1, Kt, Nw, act);
    else
      gemm_wmma_kernel<2><<<g, 32, 0, stream>>>(A0, A1, rbfp, eap, colp, Bp, bias, resid, C,
                                                M, K0, K1, Kt, Nw, act);
  };

  // geometry
  geom_kernel<<<(EE + 255) / 256, 256, 0, stream>>>(X, col, unitB, rbfB);
  // v = 0
  zero_kernel<<<(NN * HH * 3 + 255) / 256, 256, 0, stream>>>(va, NN * HH * 3);
  // node embedding: h = silu(na @ Win1 + b1) @ Win2 + b2
  gemm(0, node_attr, nullptr, nullptr, nullptr, nullptr, pWin1, b_in1, nullptr, hxB,
       NN, 64, 0, 2, HH, 1);
  gemm(0, hxB, nullptr, nullptr, nullptr, nullptr, pWin2, b_in2, nullptr, hB,
       NN, HH, 0, 4, HH, 0);

  float* vcur = va;
  float* vnext = vb;
  for (int i = 0; i < LL; ++i) {
    // edge MLP (fused gather A): t1 = silu(m_in @ We1 + be1); m = t1 @ We2 + be2
    gemm(1, hB, nullptr, rbfB, edge_attr, col, pWe1 + i * psize(128, 10), be1 + i * HH,
         nullptr, ebuf1, EE, 0, 0, 10, HH, 1);
    gemm(0, ebuf1, nullptr, nullptr, nullptr, nullptr, pWe2 + i * psize(128, 4), be2 + i * HH,
         nullptr, ebuf0, EE, HH, 0, 4, HH, 0);
    // mv = m @ Wv ; mvv = m @ Wvv
    gemm(0, ebuf0, nullptr, nullptr, nullptr, nullptr, pWv + i * psize(128, 4), nullptr,
         nullptr, ebuf1, EE, HH, 0, 4, HH, 0);
    gemm(0, ebuf0, nullptr, nullptr, nullptr, nullptr, pWvv + i * psize(128, 4), nullptr,
         nullptr, ebuf2, EE, HH, 0, 4, HH, 0);
    // agg over row (contiguous 9-wide)
    agg_kernel<<<(NN * HH + 255) / 256, 256, 0, stream>>>(ebuf0, aggB);
    // v update (uses old v incl. v[col])
    v_update_kernel<<<(NN * HH + 255) / 256, 256, 0, stream>>>(vcur, ebuf1, ebuf2, unitB,
                                                               col, vnext);
    { float* t = vcur; vcur = vnext; vnext = t; }
    // h_intra = h + silu(cat(h,agg) @ Wh1 + bh1) @ Wh2 + bh2
    gemm(2, hB, aggB, nullptr, nullptr, nullptr, pWh1 + i * psize(128, 8), bh1 + i * HH,
         nullptr, hxB, NN, HH, HH, 8, HH, 1);
    gemm(0, hxB, nullptr, nullptr, nullptr, nullptr, pWh2 + i * psize(128, 4), bh2 + i * HH,
         hB, hiB, NN, HH, 0, 4, HH, 0);
    // inter-segment attention (segments are contiguous 512-node blocks)
    seg_sum_kernel<<<NSEG, HH, 0, stream>>>(hiB, segsumB);
    score_kernel<<<(NN + 255) / 256, 256, 0, stream>>>(hiB, segsumB, scoreB);
    softstat_kernel<<<NSEG, 256, 0, stream>>>(scoreB, smaxB, sumexB);
    gated_kernel<<<(NN * HH + 255) / 256, 256, 0, stream>>>(hiB, scoreB, smaxB, sumexB, gatedB);
    // h = h_intra + silu(gated @ Wi1 + bi1) @ Wi2 + bi2
    gemm(0, gatedB, nullptr, nullptr, nullptr, nullptr, pWi1 + i * psize(128, 4), bi1 + i * HH,
         nullptr, hxB, NN, HH, 0, 4, HH, 1);
    gemm(0, hxB, nullptr, nullptr, nullptr, nullptr, pWi2 + i * psize(128, 4), bi2 + i * HH,
         hiB, hB, NN, HH, 0, 4, HH, 0);
  }

  // readout
  vt_kernel<<<(NN * HH * 3 + 255) / 256, 256, 0, stream>>>(vcur, vtB);
  gemm(0, vtB, nullptr, nullptr, nullptr, nullptr, pWgv1, nullptr, nullptr, v1tB,
       NN * 3, HH, 0, 4, HH, 0);
  gemm(0, vtB, nullptr, nullptr, nullptr, nullptr, pWgv2, nullptr, nullptr, v2tB,
       NN * 3, HH, 0, 4, HH, 0);
  v2n_kernel<<<(NN * HH + 255) / 256, 256, 0, stream>>>(v2tB, v2nB);
  gemm(2, hB, v2nB, nullptr, nullptr, nullptr, pWg1, bg1, nullptr, hxB,
       NN, HH, HH, 8, HH, 1);
  gemm(0, hxB, nullptr, nullptr, nullptr, nullptr, pWg2, bg2, nullptr, mixB,
       NN, HH, 0, 4, 256, 0);
  pv_kernel<<<(NN * 27 + 255) / 256, 256, 0, stream>>>(v1tB, mixB, Wf, pvB);
  pred_kernel<<<NSEG, 27, 0, stream>>>(pvB, predB);
  out_kernel<<<NSEG, 12, 0, stream>>>(predB, (float*)d_out);
}
